// CtxAttention_67087389164063
// MI455X (gfx1250) — matrix operand order repaired and checked by reading, not compiled, not run on hardware
//
#include <hip/hip_runtime.h>
#include <math.h>

#define N_BATCH 32
#define TI      2000
#define DIM     512          // ENC_DIM = DEC_DIM = ATT_DIM
#define TTILE   128
#define ENC_STRIDE 520       // halves per LDS row (512 + 8 pad -> spreads banks)
#define KCHUNK  64           // k-width of W LDS slab
#define W_STRIDE   72        // halves per LDS row (64 + 8 pad)
#define EPAD    2048         // padded Ti stride for score scratch

typedef __attribute__((ext_vector_type(16))) _Float16 v16h;
typedef __attribute__((ext_vector_type(8)))  _Float16 v8h;
typedef __attribute__((ext_vector_type(4)))  _Float16 v4h;
typedef __attribute__((ext_vector_type(8)))  float    v8f;

// ---------------------------------------------------------------------------
// K1: s[n,a] = b_enc[a] + sum_d W_dec[a,d] * dec_prev[n,d]
// ---------------------------------------------------------------------------
__global__ __launch_bounds__(256) void k_dec_proj(const float* __restrict__ dec_prev,
                                                  const float* __restrict__ W_dec,
                                                  const float* __restrict__ b_enc,
                                                  float* __restrict__ sBuf) {
    const int n = blockIdx.x, tid = threadIdx.x;
    __shared__ float dSh[DIM];
    for (int i = tid; i < DIM; i += 256) dSh[i] = dec_prev[n * DIM + i];
    __syncthreads();
    const float4* d4 = (const float4*)dSh;
    for (int a = tid; a < DIM; a += 256) {
        const float4* w4 = (const float4*)(W_dec + (size_t)a * DIM);
        float acc = 0.f;
#pragma unroll 8
        for (int k = 0; k < DIM / 4; ++k) {
            float4 w = w4[k], d = d4[k];
            acc += w.x * d.x + w.y * d.y + w.z * d.z + w.w * d.w;
        }
        sBuf[n * DIM + a] = acc + b_enc[a];
    }
}

// ---------------------------------------------------------------------------
// K2: e[n,t] = sum_a w_v[a] * tanh( (enc_pad[n,t,:] @ W_enc[a,:]) + s[n,a] )
//     via v_wmma_f32_16x16x32_f16. Block = (n, 128-row t-tile), 8 waves.
//     W_enc chunks register-prefetched across phases; B-fragments preloaded
//     in a burst so the WMMA stream doesn't stall on per-op dscnt waits.
// ---------------------------------------------------------------------------
__global__ __launch_bounds__(256) void k_score_wmma(const float* __restrict__ enc,
                                                    const int*   __restrict__ enc_len,
                                                    const float* __restrict__ W_enc,
                                                    const float* __restrict__ w_v,
                                                    const float* __restrict__ sBuf,
                                                    float* __restrict__ eBuf) {
    __shared__ __attribute__((aligned(16))) _Float16 encSh[TTILE * ENC_STRIDE]; // 130 KB
    __shared__ __attribute__((aligned(16))) _Float16 wSh[128 * W_STRIDE];       // 18 KB
    __shared__ float sSh[DIM];
    __shared__ float wvSh[DIM];

    const int tid   = threadIdx.x;
    const int n     = blockIdx.y;
    const int tBase = blockIdx.x * TTILE;
    const int lane  = tid & 31;
    const int wvid  = tid >> 5;      // wave id 0..7
    const int lr    = lane & 15;
    const int half  = lane >> 4;     // 0: lanes 0-15, 1: lanes 16-31
    const int wRow  = 16 * wvid;     // wave's t-row base within tile

    // stage s = dec_part + b_enc, and w_v
    for (int i = tid; i < DIM; i += 256) { sSh[i] = sBuf[n * DIM + i]; wvSh[i] = w_v[i]; }

    // ---- stage enc tile fp32 -> f16, 16 loads in flight per batch ----
    const float4* encG = (const float4*)(enc + ((size_t)n * TI + tBase) * DIM);
    const int maxRow = (TI - tBase - 1 < TTILE - 1) ? (TI - tBase - 1) : (TTILE - 1);
    for (int ii = 0; ii < 64; ii += 16) {
        float4 tmp[16];
#pragma unroll
        for (int j = 0; j < 16; ++j) {
            const int i  = (ii + j) * 256 + tid;
            const int tl = i >> 7;                 // 128 float4 per row
            const int kq = i & 127;
            const int tc = tl < maxRow ? tl : maxRow;  // clamp -> always valid addr
            tmp[j] = encG[(size_t)tc * 128 + kq];
        }
#pragma unroll
        for (int j = 0; j < 16; ++j) {
            const int i  = (ii + j) * 256 + tid;
            const int tl = i >> 7;
            const int kq = i & 127;
            float4 v = tmp[j];
            if (tl > maxRow) v = make_float4(0.f, 0.f, 0.f, 0.f);
            v4h h; h[0] = (_Float16)v.x; h[1] = (_Float16)v.y;
                   h[2] = (_Float16)v.z; h[3] = (_Float16)v.w;
            *(v4h*)&encSh[tl * ENC_STRIDE + kq * 4] = h;
        }
    }

    // ---- W_enc chunk prefetch machinery: chunk p = (ag = p>>3, kc = p&7) ----
    // chunk = rows ag*128..+127, cols kc*64..+63 ; 2048 float4 -> 8 per thread
    float4 wreg[8];
    const float4* wG = (const float4*)W_enc;      // [512][128] float4
    auto issueW = [&](int p) {
        const int ag = p >> 3, kc = p & 7;
#pragma unroll
        for (int j = 0; j < 8; ++j) {
            const int i  = j * 256 + tid;
            const int al = i >> 4;                // 16 float4 per 64-wide row
            const int kq = i & 15;
            wreg[j] = wG[(size_t)(ag * 128 + al) * 128 + kc * 16 + kq];
        }
    };
    auto storeW = [&]() {
#pragma unroll
        for (int j = 0; j < 8; ++j) {
            const int i  = j * 256 + tid;
            const int al = i >> 4;
            const int kq = i & 15;
            float4 v = wreg[j];
            v4h h; h[0] = (_Float16)v.x; h[1] = (_Float16)v.y;
                   h[2] = (_Float16)v.z; h[3] = (_Float16)v.w;
            *(v4h*)&wSh[al * W_STRIDE + kq * 4] = h;
        }
    };

    issueW(0);

    float eacc[8];
#pragma unroll
    for (int r = 0; r < 8; ++r) eacc[r] = 0.f;

    int p = 0;
    for (int ag = 0; ag < 4; ++ag) {             // a in groups of 128
        v8f c[8];
#pragma unroll
        for (int at = 0; at < 8; ++at)
#pragma unroll
            for (int j = 0; j < 8; ++j) c[at][j] = 0.f;

        for (int kc = 0; kc < 8; ++kc, ++p) {    // k in chunks of 64
            __syncthreads();                     // WAR: prev readers of wSh done
            storeW();                            // current chunk regs -> LDS (f16)
            if (p + 1 < 32) issueW(p + 1);       // prefetch next chunk (in flight
                                                 // across the WMMA phase below)
            __syncthreads();                     // RAW: wSh visible

#pragma unroll
            for (int ks = 0; ks < 2; ++ks) {     // two 32-wide sub-chunks
                // A fragment: lane<16 -> M=lr, K {0..7,16..23}; lane>=16 -> {8..15,24..31}
                const _Float16* ap =
                    &encSh[(wRow + lr) * ENC_STRIDE + kc * KCHUNK + ks * 32 + 8 * half];
                v8h a0 = *(const v8h*)ap;
                v8h a1 = *(const v8h*)(ap + 16);
                v16h afrag = __builtin_shufflevector(a0, a1,
                                 0,1,2,3,4,5,6,7,8,9,10,11,12,13,14,15);

                // burst-load all 8 B fragments (16 ds_load_b128 in flight),
                // then issue the 8 WMMAs back-to-back
                v16h bf[8];
#pragma unroll
                for (int at = 0; at < 8; ++at) {
                    // B fragment: lane<16 -> N=lr, K {0..7,16..23}; lane>=16 -> {8..15,24..31}
                    const _Float16* bp =
                        &wSh[(at * 16 + lr) * W_STRIDE + ks * 32 + 8 * half];
                    v8h b0 = *(const v8h*)bp;
                    v8h b1 = *(const v8h*)(bp + 16);
                    bf[at] = __builtin_shufflevector(b0, b1,
                                 0,1,2,3,4,5,6,7,8,9,10,11,12,13,14,15);
                }
#pragma unroll
                for (int at = 0; at < 8; ++at)
                    c[at] = __builtin_amdgcn_wmma_f32_16x16x32_f16(
                                false, afrag, false, bf[at], (short)0, c[at],
                                false, false);
            }
        }

        // epilogue: C layout -> VGPR r is row M = r + 8*half, col N = lr
#pragma unroll
        for (int at = 0; at < 8; ++at) {
            const int aIdx = ag * 128 + at * 16 + lr;
            const float sv = sSh[aIdx];
            const float ww = wvSh[aIdx];
#pragma unroll
            for (int r = 0; r < 8; ++r)
                eacc[r] += tanhf(c[at][r] + sv) * ww;
        }
    }

    // reduce over the 16 columns held across each 16-lane half, write e
    const int encLenN = enc_len[n];
#pragma unroll
    for (int r = 0; r < 8; ++r) {
        float v = eacc[r];
        v += __shfl_xor(v, 1);
        v += __shfl_xor(v, 2);
        v += __shfl_xor(v, 4);
        v += __shfl_xor(v, 8);
        const int t = tBase + wRow + r + 8 * half;
        if (lr == 0 && t < TI)
            eBuf[n * EPAD + t] = (t >= encLenN) ? -__builtin_inff() : v;
    }
}

// ---------------------------------------------------------------------------
// K3: masked softmax over t per batch row
// ---------------------------------------------------------------------------
__global__ __launch_bounds__(256) void k_softmax(const float* __restrict__ eBuf,
                                                 float* __restrict__ ali) {
    const int n = blockIdx.x, tid = threadIdx.x;
    __shared__ float red[256];
    const float* e = eBuf + n * EPAD;
    float m = -__builtin_inff();
    for (int t = tid; t < TI; t += 256) m = fmaxf(m, e[t]);
    red[tid] = m; __syncthreads();
    for (int s = 128; s > 0; s >>= 1) {
        if (tid < s) red[tid] = fmaxf(red[tid], red[tid + s]);
        __syncthreads();
    }
    const float mx = red[0];
    __syncthreads();
    float* out = ali + (size_t)n * TI;
    float sum = 0.f;
    for (int t = tid; t < TI; t += 256) {
        const float pv = expf(e[t] - mx);  // exp(-inf)=0 handles mask
        out[t] = pv;
        sum += pv;
    }
    red[tid] = sum; __syncthreads();
    for (int s = 128; s > 0; s >>= 1) {
        if (tid < s) red[tid] += red[tid + s];
        __syncthreads();
    }
    const float inv = 1.f / red[0];
    for (int t = tid; t < TI; t += 256) out[t] *= inv;
}

// ---------------------------------------------------------------------------
// K4: ctx[n,d] = sum_t ali[n,t] * enc[n,t,d]   (HBM-bound, coalesced over d)
// ---------------------------------------------------------------------------
__global__ __launch_bounds__(256) void k_ctx(const float* __restrict__ enc,
                                             const float* __restrict__ ali,
                                             float* __restrict__ ctx) {
    __shared__ float aliSh[TI];
    const int n = blockIdx.y, tid = threadIdx.x;
    const int d = blockIdx.x * 256 + tid;
    for (int t = tid; t < TI; t += 256) aliSh[t] = ali[(size_t)n * TI + t];
    __syncthreads();
    const float* encN = enc + (size_t)n * TI * DIM + d;
    float acc = 0.f;
#pragma unroll 8
    for (int t = 0; t < TI; ++t) acc += aliSh[t] * encN[(size_t)t * DIM];
    ctx[n * DIM + d] = acc;
}

// ---------------------------------------------------------------------------
extern "C" void kernel_launch(void* const* d_in, const int* in_sizes, int n_in,
                              void* d_out, int out_size, void* d_ws, size_t ws_size,
                              hipStream_t stream) {
    const float* enc_pad  = (const float*)d_in[0];
    const int*   enc_len  = (const int*)  d_in[1];
    const float* dec_prev = (const float*)d_in[2];
    // d_in[3] = ali_prev (unused by reference)
    const float* W_enc    = (const float*)d_in[4];
    const float* b_enc    = (const float*)d_in[5];
    const float* W_dec    = (const float*)d_in[6];
    const float* w_v      = (const float*)d_in[7];

    float* ali = (float*)d_out;                     // N*TI
    float* ctx = ali + (size_t)N_BATCH * TI;        // N*DIM

    float* sBuf = (float*)d_ws;                     // N*DIM
    float* eBuf = sBuf + N_BATCH * DIM;             // N*EPAD

    k_dec_proj  <<<dim3(N_BATCH),      dim3(256), 0, stream>>>(dec_prev, W_dec, b_enc, sBuf);
    k_score_wmma<<<dim3(16, N_BATCH),  dim3(256), 0, stream>>>(enc_pad, enc_len, W_enc, w_v, sBuf, eBuf);
    k_softmax   <<<dim3(N_BATCH),      dim3(256), 0, stream>>>(eBuf, ali);
    k_ctx       <<<dim3(2, N_BATCH),   dim3(256), 0, stream>>>(enc_pad, ali, ctx);
}
